// ANINetwork_47880295416070
// MI455X (gfx1250) — matrix operand order repaired and checked by reading, not compiled, not run on hardware
//
#include <hip/hip_runtime.h>
#include <hip/hip_bf16.h>

// ---------------------------------------------------------------------------
// Types for CDNA5 WMMA (wave32): v_wmma_f32_16x16x32_bf16
// ---------------------------------------------------------------------------
typedef __attribute__((ext_vector_type(16))) __bf16        v16bf;
typedef __attribute__((ext_vector_type(8)))  __bf16        v8bf;
typedef __attribute__((ext_vector_type(8)))  float         v8f;
typedef __attribute__((ext_vector_type(4)))  float         v4f;
typedef __attribute__((ext_vector_type(4)))  unsigned int  v4u;
typedef __attribute__((ext_vector_type(2)))  unsigned int  v2u;

union FragA { v16bf v; v8bf h[2]; };   // A: 16x32 bf16 (ISA 7.12.2 16-bit A layout)
union FragB { v16bf v; v4u  q[2]; };   // B: 32x16 bf16 (packed to match lane layout)

// round-to-nearest-even f32 -> bf16 (bit pattern as ushort)
static __device__ __forceinline__ unsigned int f2bf(float f) {
  unsigned int u = __float_as_uint(f);
  u += 0x7FFFu + ((u >> 16) & 1u);
  return u >> 16;
}

static __device__ __forceinline__ float celu1(float x) {
  return (x > 0.0f) ? x : (__expf(x) - 1.0f);
}

// ---------------------------------------------------------------------------
// Species bucketing
// ---------------------------------------------------------------------------
__global__ void k_init(int* __restrict__ meta, int* __restrict__ perm, int permLen) {
  int i = blockIdx.x * blockDim.x + threadIdx.x;
  if (i < 16) meta[i] = 0;                 // counts[0..3], cursors[4..7], base[8..12]
  if (i < permLen) perm[i] = -1;
}

__global__ void k_count(const int* __restrict__ sp, int* __restrict__ counts, int n) {
  int i = blockIdx.x * blockDim.x + threadIdx.x;
  if (i < n) atomicAdd(&counts[sp[i]], 1);
}

__global__ void k_prefix(const int* __restrict__ counts, int* __restrict__ base) {
  if (blockIdx.x == 0 && threadIdx.x == 0) {
    int acc = 0;
    for (int s = 0; s < 4; ++s) {
      base[s] = acc;
      acc += ((counts[s] + 127) / 128) * 128;   // pad each bucket to one block (8 tiles)
    }
    base[4] = acc;
  }
}

__global__ void k_scatter(const int* __restrict__ sp, const int* __restrict__ base,
                          int* __restrict__ cursors, int* __restrict__ perm, int n) {
  int i = blockIdx.x * blockDim.x + threadIdx.x;
  if (i < n) {
    int s = sp[i];
    int pos = base[s] + atomicAdd(&cursors[s], 1);
    perm[pos] = i;
  }
}

// ---------------------------------------------------------------------------
// Pack f32 weights -> bf16 in WMMA B-fragment layout.
// Fragment (nt,kt): 32 lanes x 16 bf16 contiguous (1024 B).
//   lane l, elem e  ->  K = kt*32 + (l>>4)*8 + (e&7) + (e>>3)*16,  N = nt*16 + (l&15)
// ---------------------------------------------------------------------------
__global__ void k_pack(const float* __restrict__ W, unsigned short* __restrict__ dst,
                       int din, int dout, int nT, int kT) {
  int idx = blockIdx.x * blockDim.x + threadIdx.x;
  int total = 4 * nT * kT * 512;
  if (idx >= total) return;
  int e  = idx & 15;
  int l  = (idx >> 4) & 31;
  int f  = idx >> 9;
  int kt = f % kT; f /= kT;
  int nt = f % nT; f /= nT;
  int s  = f;
  int K  = kt * 32 + (l >> 4) * 8 + (e & 7) + (e >> 3) * 16;
  int Nc = nt * 16 + (l & 15);
  float v = (Nc < dout) ? W[((size_t)s * din + K) * (size_t)dout + Nc] : 0.0f;
  dst[idx] = (unsigned short)f2bf(v);
}

// ---------------------------------------------------------------------------
// One MLP layer on a 16-atom tile (one wave).  in/out are LDS bf16 row-major
// [16][DIN] / [16][DOUT].  Weights are packed fragments in global (L2).
// ---------------------------------------------------------------------------
template <int DIN, int DOUT, bool ACT>
static __device__ __forceinline__ void run_layer(const unsigned short* __restrict__ in,
                                                 unsigned short* __restrict__ out,
                                                 const unsigned short* __restrict__ w,
                                                 const float* __restrict__ bias,
                                                 int lane) {
  constexpr int KT = DIN / 32;
  constexpr int NT = DOUT / 16;
  const int hi  = lane >> 4;
  const int col = lane & 15;          // also the A-matrix row (m) this lane owns
  for (int nt = 0; nt < NT; ++nt) {
    float bv = bias[nt * 16 + col];
    v8f acc;
#pragma unroll
    for (int r = 0; r < 8; ++r) acc[r] = bv;
#pragma unroll
    for (int kt = 0; kt < KT; ++kt) {
      FragA a;
      const unsigned short* arow = in + col * DIN + kt * 32 + hi * 8;
      a.h[0] = *(const v8bf*)(arow);
      a.h[1] = *(const v8bf*)(arow + 16);
      const unsigned short* wp = w + (((size_t)(nt * KT + kt)) << 9) + lane * 16;
      FragB b;
      b.q[0] = *(const v4u*)(wp);
      b.q[1] = *(const v4u*)(wp + 8);
      if (kt + 1 < KT) __builtin_prefetch(wp + 512, 0, 1);   // global_prefetch_b8
      acc = __builtin_amdgcn_wmma_f32_16x16x32_bf16(
          /*neg_a=*/false, a.v, /*neg_b=*/false, b.v,
          /*c_mod=*/(short)0, acc, /*reuse_a=*/false, /*reuse_b=*/false);
    }
#pragma unroll
    for (int r = 0; r < 8; ++r) {
      float x = acc[r];
      if (ACT) x = celu1(x);
      out[(r + 8 * hi) * DOUT + nt * 16 + col] = (unsigned short)f2bf(x);
    }
  }
}

// ---------------------------------------------------------------------------
// Main kernel: 8 waves/block, one 16-atom species-uniform tile per wave.
// ---------------------------------------------------------------------------
#define TILE_LDS (16 * 384 + 16 * 160)   // ushorts per wave (bufX + bufH)

__global__ __launch_bounds__(256) void k_mlp(const int* __restrict__ perm,
                                             const int* __restrict__ base,
                                             const float* __restrict__ aev,
                                             const unsigned short* __restrict__ wpack,
                                             const float* __restrict__ b1,
                                             const float* __restrict__ b2,
                                             const float* __restrict__ b3,
                                             const float* __restrict__ b4,
                                             float* __restrict__ eAtom) {
  extern __shared__ unsigned short lds[];
  const int lane = threadIdx.x & 31;
  const int wave = threadIdx.x >> 5;
  const int hi   = lane >> 4;
  const int col  = lane & 15;

  unsigned short* bufX = lds + wave * TILE_LDS;   // 16x384 (X, later H2 @128)
  unsigned short* bufH = bufX + 16 * 384;         // 16x160 (H1, later H3 @96)

  const long tileBase = (long)blockIdx.x * 128 + (long)wave * 16;

  // species of this tile (buckets are 128-atom aligned; tail region -> s=3, all pads)
  int s = 0;
  if (tileBase >= base[1]) s = 1;
  if (tileBase >= base[2]) s = 2;
  if (tileBase >= base[3]) s = 3;

  // ---- stage X tile: gather aev rows via perm, f32 -> bf16 into LDS ----
  // Vectorized: one b128 global load + one b64 LDS store per 4 elements.
  for (int r = 0; r < 16; ++r) {
    int ai = perm[tileBase + r];
#pragma unroll
    for (int j = 0; j < 3; ++j) {
      int c = j * 128 + lane * 4;
      v4f x = {0.0f, 0.0f, 0.0f, 0.0f};
      if (ai >= 0) x = *(const v4f*)(aev + (size_t)ai * 384 + c);
      v2u p;
      p.x = f2bf(x.x) | (f2bf(x.y) << 16);
      p.y = f2bf(x.z) | (f2bf(x.w) << 16);
      *(v2u*)(bufX + r * 384 + c) = p;
    }
  }

  // ---- per-species packed weight bases ----
  const unsigned short* w1 = wpack + (size_t)s * (10 * 12 * 512);
  const unsigned short* w2 = wpack + 4 * (10 * 12 * 512) + (size_t)s * (8 * 5 * 512);
  const unsigned short* w3 = wpack + 4 * (10 * 12 * 512) + 4 * (8 * 5 * 512) + (size_t)s * (6 * 4 * 512);
  const unsigned short* w4 = wpack + 4 * (10 * 12 * 512) + 4 * (8 * 5 * 512) + 4 * (6 * 4 * 512) + (size_t)s * (1 * 3 * 512);

  run_layer<384, 160, true>(bufX, bufH, w1, b1 + s * 160, lane);   // X  -> H1
  run_layer<160, 128, true>(bufH, bufX, w2, b2 + s * 128, lane);   // H1 -> H2
  run_layer<128,  96, true>(bufX, bufH, w3, b3 + s *  96, lane);   // H2 -> H3

  // ---- layer 4: 96 -> 1 (output tile padded to 16 cols; only col 0 valid) ----
  {
    float bv = (col == 0) ? b4[s] : 0.0f;
    v8f acc;
#pragma unroll
    for (int r = 0; r < 8; ++r) acc[r] = bv;
#pragma unroll
    for (int kt = 0; kt < 3; ++kt) {
      FragA a;
      const unsigned short* arow = bufH + col * 96 + kt * 32 + hi * 8;
      a.h[0] = *(const v8bf*)(arow);
      a.h[1] = *(const v8bf*)(arow + 16);
      const unsigned short* wp = w4 + (((size_t)kt) << 9) + lane * 16;
      FragB b;
      b.q[0] = *(const v4u*)(wp);
      b.q[1] = *(const v4u*)(wp + 8);
      acc = __builtin_amdgcn_wmma_f32_16x16x32_bf16(false, a.v, false, b.v,
                                                    (short)0, acc, false, false);
    }
    if (col == 0) {   // lanes 0 and 16 hold column 0 for rows r+8*hi
#pragma unroll
      for (int r = 0; r < 8; ++r) {
        int m  = r + 8 * hi;
        int ai = perm[tileBase + m];
        if (ai >= 0) eAtom[ai] = acc[r];
      }
    }
  }
}

// ---------------------------------------------------------------------------
// Deterministic fixed-order per-molecule reduction (64 atoms each)
// ---------------------------------------------------------------------------
__global__ void k_reduce(const float* __restrict__ e, float* __restrict__ out, int A) {
  __shared__ float sm[64];
  int t = threadIdx.x;
  sm[t] = e[(size_t)blockIdx.x * A + t];
  __syncthreads();
  for (int w = 32; w > 0; w >>= 1) {
    if (t < w) sm[t] += sm[t + w];
    __syncthreads();
  }
  if (t == 0) out[blockIdx.x] = sm[0];
}

// ---------------------------------------------------------------------------
extern "C" void kernel_launch(void* const* d_in, const int* in_sizes, int n_in,
                              void* d_out, int out_size, void* d_ws, size_t ws_size,
                              hipStream_t stream) {
  const int*   species = (const int*)  d_in[0];
  const float* aev     = (const float*)d_in[1];
  const float* W1 = (const float*)d_in[2]; const float* b1 = (const float*)d_in[3];
  const float* W2 = (const float*)d_in[4]; const float* b2 = (const float*)d_in[5];
  const float* W3 = (const float*)d_in[6]; const float* b3 = (const float*)d_in[7];
  const float* W4 = (const float*)d_in[8]; const float* b4 = (const float*)d_in[9];

  const int N  = in_sizes[0];   // B*A atoms
  const int Bm = out_size;      // molecules
  const int A  = N / Bm;        // atoms per molecule (64)

  // ---- workspace partition (all 256 B aligned) ----
  char* ws = (char*)d_ws;
  int* meta    = (int*)ws;            // 16 ints: counts[4], cursors[4], base[5]
  int* counts  = meta;
  int* cursors = meta + 4;
  int* base    = meta + 8;
  int  permLen = N + 4 * 128;
  int* perm    = (int*)(ws + 256);
  size_t off   = 256 + (((size_t)permLen * 4 + 255) & ~(size_t)255);
  float* eAtom = (float*)(ws + off);
  off += (((size_t)N * 4 + 255) & ~(size_t)255);
  unsigned short* wpack = (unsigned short*)(ws + off);

  // packed-weight sub-offsets (ushorts)
  const int L1 = 4 * 10 * 12 * 512;   // 245760
  const int L2 = 4 *  8 *  5 * 512;   //  81920
  const int L3 = 4 *  6 *  4 * 512;   //  49152
  const int L4 = 4 *  1 *  3 * 512;   //   6144

  // ---- bucketing ----
  k_init<<<(permLen + 255) / 256, 256, 0, stream>>>(meta, perm, permLen);
  k_count<<<(N + 255) / 256, 256, 0, stream>>>(species, counts, N);
  k_prefix<<<1, 1, 0, stream>>>(counts, base);
  k_scatter<<<(N + 255) / 256, 256, 0, stream>>>(species, base, cursors, perm, N);

  // ---- pack weights to bf16 fragment layout (L2-resident, ~0.75 MB) ----
  k_pack<<<(L1 + 255) / 256, 256, 0, stream>>>(W1, wpack,                384, 160, 10, 12);
  k_pack<<<(L2 + 255) / 256, 256, 0, stream>>>(W2, wpack + L1,           160, 128,  8,  5);
  k_pack<<<(L3 + 255) / 256, 256, 0, stream>>>(W3, wpack + L1 + L2,      128,  96,  6,  4);
  k_pack<<<(L4 + 255) / 256, 256, 0, stream>>>(W4, wpack + L1 + L2 + L3,  96,   1,  1,  3);

  // ---- main WMMA MLP: 8 waves/block, 128 atoms/block ----
  int nBlocks = (N + 127) / 128 + 4;            // covers worst-case bucket padding
  size_t shmem = (size_t)8 * TILE_LDS * sizeof(unsigned short);   // 139264 B
  k_mlp<<<nBlocks, 256, shmem, stream>>>(perm, base, aev, wpack,
                                         b1, b2, b3, b4, eAtom);

  // ---- per-molecule sum ----
  k_reduce<<<Bm, 64, 0, stream>>>(eAtom, (float*)d_out, A);
}
